// Embedding2ScoreSAN_46239617909198
// MI455X (gfx1250) — matrix operand order repaired and checked by reading, not compiled, not run on hardware
//
#include <hip/hip_runtime.h>
#include <hip/hip_bf16.h>

// ---------------------------------------------------------------------------
// Problem constants (from the reference)
// ---------------------------------------------------------------------------
#define D 100
#define SAN_BLOCKS 3
#define NUM_GRAPHS 2048
#define NODES_PER_GRAPH 50
#define VOCAB 50257
#define KPAD 128                 // K=100 padded to 128 for 4x WMMA k=32
#define NTILES 3142              // ceil(50257/16)
#define NPAD (NTILES * 16)       // 50272 padded item rows

typedef __attribute__((ext_vector_type(16))) _Float16 v16h;
typedef __attribute__((ext_vector_type(8)))  _Float16 v8h;
typedef __attribute__((ext_vector_type(8)))  float    v8f;

// ---------------------------------------------------------------------------
// Kernel 1: per-graph SAN chain on the LAST node only, fused with W3.
// One 128-thread block per graph; threads 0..99 each own one output element.
// Writes s_h as f16, padded to KPAD with zeros.
// ---------------------------------------------------------------------------
__global__ void san_chain_kernel(const float* __restrict__ node,
                                 const float* __restrict__ Wv, const float* __restrict__ bv,
                                 const float* __restrict__ Wo, const float* __restrict__ bo,
                                 const float* __restrict__ d1w, const float* __restrict__ d1b,
                                 const float* __restrict__ d2w, const float* __restrict__ d2b,
                                 const float* __restrict__ W3w, const float* __restrict__ W3b,
                                 _Float16* __restrict__ sh16) {
    __shared__ float x[D], t[D], a[D], h[D], vn[D];
    const int g = blockIdx.x;
    const int i = threadIdx.x;
    const long node_idx = (long)g * NODES_PER_GRAPH + (NODES_PER_GRAPH - 1);

    if (i < D) {
        float v = node[node_idx * D + i];
        x[i] = v;
        vn[i] = v;
    }
    __syncthreads();

    for (int blk = 0; blk < SAN_BLOCKS; ++blk) {
        if (i < D) {                       // t = Wv x + bv
            float s = bv[i];
            for (int j = 0; j < D; ++j) s += Wv[i * D + j] * x[j];
            t[i] = s;
        }
        __syncthreads();
        if (i < D) {                       // a = Wo t + bo
            float s = bo[i];
            for (int j = 0; j < D; ++j) s += Wo[i * D + j] * t[j];
            a[i] = s;
        }
        __syncthreads();
        if (i < D) {                       // h = relu(d1w a + d1b)
            float s = d1b[i];
            for (int j = 0; j < D; ++j) s += d1w[i * D + j] * a[j];
            h[i] = fmaxf(s, 0.0f);
        }
        __syncthreads();
        if (i < D) {                       // x = a + d2w h + d2b
            float s = d2b[i];
            for (int j = 0; j < D; ++j) s += d2w[i * D + j] * h[j];
            x[i] = a[i] + s;
        }
        __syncthreads();
    }

    // s_h[i] = W3b[i] + sum_j W3w[i][j]*vn[j] + sum_j W3w[i][100+j]*x[j]
    float sh = 0.0f;
    if (i < D) {
        sh = W3b[i];
        for (int j = 0; j < D; ++j) sh += W3w[i * (2 * D) + j] * vn[j];
        for (int j = 0; j < D; ++j) sh += W3w[i * (2 * D) + D + j] * x[j];
    }
    if (i < KPAD)
        sh16[(size_t)g * KPAD + i] = (i < D) ? (_Float16)sh : (_Float16)0.0f;
}

// ---------------------------------------------------------------------------
// Kernel 2: item_table (f32, [VOCAB,100]) -> f16 padded [NPAD, KPAD]
// Stays resident in the 192 MB L2 for the GEMM (12.9 MB).
// ---------------------------------------------------------------------------
__global__ void cvt_item_kernel(const float* __restrict__ item,
                                _Float16* __restrict__ it16) {
    const size_t idx = (size_t)blockIdx.x * blockDim.x + threadIdx.x;
    if (idx >= (size_t)NPAD * KPAD) return;
    const int v = (int)(idx >> 7);     // row
    const int k = (int)(idx & 127);    // padded K
    float val = 0.0f;
    if (v < VOCAB && k < D) val = item[(size_t)v * D + k];
    it16[idx] = (_Float16)val;
}

// ---------------------------------------------------------------------------
// Kernel 3: z[g,v] = sum_k s_h[g,k] * item[v,k]   (M=2048, N=50257, K=100->128)
// One wave (32 lanes) per 16x16 output tile; 4x v_wmma_f32_16x16x32_f16.
//
// wave32 fragment maps as loaded here (contiguous k order, same for A and B):
//   A (16x32 f16): lane L -> row M=L&15; elems 0..15 = K(32c + 16*(L>>4) + 0..15)
//   B (32x16 f16): lane L -> col N=L&15; elems 0..15 = K(32c + 16*(L>>4) + 0..15)
//   C/D (16x16 f32): lane L, vgpr r -> row M = r + 8*(L>>4), col N = L&15
// ---------------------------------------------------------------------------
__global__ void gemm_wmma_kernel(const _Float16* __restrict__ sh16,
                                 const _Float16* __restrict__ it16,
                                 float* __restrict__ out) {
    const int lane = threadIdx.x & 31;
    const int wave = threadIdx.x >> 5;
    const int tile = blockIdx.x * (blockDim.x >> 5) + wave;   // < 128*NTILES by grid

    const int mT = tile / NTILES;
    const int nT = tile - mT * NTILES;
    const int hi  = lane >> 4;     // 0 or 1
    const int col = lane & 15;     // M index for A, N index for B/C/D

    const _Float16* aBase = sh16 + (size_t)(mT * 16 + col) * KPAD;
    const _Float16* bBase = it16 + (size_t)(nT * 16 + col) * KPAD;

    v8f acc = {};
#pragma unroll
    for (int c = 0; c < 4; ++c) {
        v16h afrag = *(const v16h*)(aBase + 32 * c + 16 * hi);
        v16h bfrag = *(const v16h*)(bBase + 32 * c + 16 * hi);
        acc = __builtin_amdgcn_wmma_f32_16x16x32_f16(
                  /*neg_a=*/false, afrag, /*neg_b=*/false, bfrag,
                  /*c_mod=*/(short)0, acc, /*reuse_a=*/false, /*reuse_b=*/false);
    }

    const int n = nT * 16 + col;
    if (n < VOCAB) {
        size_t base = (size_t)(mT * 16 + 8 * hi) * VOCAB + (size_t)n;
#pragma unroll
        for (int r = 0; r < 8; ++r)
            out[base + (size_t)r * VOCAB] = acc[r];
    }
}

// ---------------------------------------------------------------------------
extern "C" void kernel_launch(void* const* d_in, const int* in_sizes, int n_in,
                              void* d_out, int out_size, void* d_ws, size_t ws_size,
                              hipStream_t stream) {
    // setup_inputs() order:
    const float* node = (const float*)d_in[0];   // [N_TOTAL, D]
    const float* item = (const float*)d_in[1];   // [VOCAB, D]
    // d_in[2] batch, d_in[3] num_count: unused (last_idx is analytic)
    const float* Wv  = (const float*)d_in[4];
    const float* bv  = (const float*)d_in[5];
    const float* Wo  = (const float*)d_in[6];
    const float* bo  = (const float*)d_in[7];
    const float* d1w = (const float*)d_in[8];
    const float* d1b = (const float*)d_in[9];
    const float* d2w = (const float*)d_in[10];
    const float* d2b = (const float*)d_in[11];
    const float* W3w = (const float*)d_in[12];   // [D, 2D]
    const float* W3b = (const float*)d_in[13];
    float* out = (float*)d_out;                  // [NUM_GRAPHS, VOCAB]

    // Workspace layout
    _Float16* it16 = (_Float16*)d_ws;                                   // NPAD*KPAD
    _Float16* sh16 = (_Float16*)((char*)d_ws + (size_t)NPAD * KPAD * 2); // 2048*KPAD

    // 1) SAN chain on last nodes + W3 fusion -> s_h (f16, K-padded)
    san_chain_kernel<<<NUM_GRAPHS, 128, 0, stream>>>(
        node, Wv, bv, Wo, bo, d1w, d1b, d2w, d2b, W3w, W3b, sh16);

    // 2) item_table -> f16, padded [NPAD, KPAD]
    {
        const size_t total = (size_t)NPAD * KPAD;
        const int threads = 256;
        const int blocks = (int)((total + threads - 1) / threads);
        cvt_item_kernel<<<blocks, threads, 0, stream>>>(item, it16);
    }

    // 3) WMMA GEMM: 128 x NTILES 16x16 tiles, 8 waves (256 threads) per block
    {
        const int tiles = (NUM_GRAPHS / 16) * NTILES;   // 402176
        const int wavesPerBlock = 8;
        const int blocks = tiles / wavesPerBlock;        // 50272, exact
        gemm_wmma_kernel<<<blocks, wavesPerBlock * 32, 0, stream>>>(sh16, it16, out);
    }
}
